// PINNMambaEncoder_65678639890530
// MI455X (gfx1250) — compile-verified, matrix operand order checked by reading, not run on hardware
//
#include <hip/hip_runtime.h>

#define NLAY 4
#define TOKENS 65536
#define TM 32
#define NBLOCKS (TOKENS / TM)   // 2048

// bf16 weight arena (transposed [N][K]) per-layer element offsets
#define OFF_WIN 0          // [512][128]
#define OFF_WXP 65536      // [2][48][256]  (N padded 40->48, pad rows zero)
#define OFF_WM1 90112      // [512][512]
#define OFF_WM2 352256     // [128][512]
#define WELEMS  417792

// LDS arena byte offsets (liveness-checked reuse)
#define SO_H    0          // bf16 [32][128]     stages A-B
#define SO_CW   8192       // f32  [2][256]      stages A-B (epilogue of B)
#define SO_CB   10240      // f32  [2][256]      stages A-B
#define SO_XA   12288      // bf16 [2][32][256]  stages B-E
#define SO_GZ   45056      // bf16 [32][256]     stages B-E
#define SO_CAT  61440      // bf16 [32][512]     stages E-F
#define SO_DBC  0          // f32  [2][32][48]   stages D-E (H/CW/CB dead)
#define SO_WDT  94208      // bf16 [2][8][256]   stages D-E
#define SO_BDT  102400     // f32  [2][256]      stages D-E
#define SO_DSK  104448     // f32  [2][256]      stages D-E
#define SO_M2   0          // bf16 [32][512]     stages F-H (DBC/XA-head dead)
#define SO_OUT  32768      // f32  [32][128]     stages H-I (XA-tail/GZ dead)
#define SMEM_BYTES 106496  // 104 KB -> 3 workgroups per WGP (320 KB LDS)

typedef __attribute__((ext_vector_type(16))) __bf16 bf16x16;
typedef __attribute__((ext_vector_type(8)))  __bf16 bf16x8;
typedef __attribute__((ext_vector_type(8)))  float  floatx8;

// Branch-free tanh-approx gelu: 0.5x(1+tanh(u)) == x / (1 + e^{-2u})
__device__ __forceinline__ float gelu_f(float x) {
  float u2 = -1.5957691216057308f * (x + 0.044715f * x * x * x);  // -2u
  return x * __builtin_amdgcn_rcpf(1.0f + __expf(u2));
}
// Branch-free stable softplus: max(x,0) + ln(1 + e^{-|x|})
__device__ __forceinline__ float softplus_f(float x) {
  return fmaxf(x, 0.0f) + __logf(1.0f + __expf(-fabsf(x)));
}

// A/B fragment for v_wmma_f32_16x16x32_bf16, source row-major [row][K]:
// lane = m (A) or n (B) = lane&15, half = lane>>4.
// elements 0..7  <- K = kbase..kbase+7      (16B contiguous)
// elements 8..15 <- K = kbase+16..kbase+23  (16B contiguous)
__device__ __forceinline__ bf16x16 load_frag(const __bf16* base, int stride, int row, int kbase) {
  const __bf16* p = base + (size_t)row * (size_t)stride + kbase;
  bf16x8 lo = *reinterpret_cast<const bf16x8*>(p);
  bf16x8 hi = *reinterpret_cast<const bf16x8*>(p + 16);
  bf16x16 r;
#pragma unroll
  for (int t = 0; t < 8; ++t) { r[t] = lo[t]; r[8 + t] = hi[t]; }
  return r;
}

// 16x16 output tile; epilogue functor epi(row, col, acc) consumes the C fragment.
// C/D layout: VGPR v holds row m = v + 8*(lane>=16), col n = lane&15.
template <int KT, class Epi>
__device__ __forceinline__ void gemm_tile(const __bf16* Ab, int as, int mbase,
                                          const __bf16* Bb, int bs, int nbase,
                                          int lane, Epi epi) {
  const int mn = lane & 15, half = lane >> 4;
  floatx8 c = {};
#pragma unroll
  for (int ks = 0; ks < KT; ++ks) {
    bf16x16 a = load_frag(Ab, as, mbase + mn, ks * 32 + half * 8);
    bf16x16 b = load_frag(Bb, bs, nbase + mn, ks * 32 + half * 8);
    c = __builtin_amdgcn_wmma_f32_16x16x32_bf16(false, a, false, b, (short)0, c, false, false);
  }
  const int col = nbase + mn;
#pragma unroll
  for (int v = 0; v < 8; ++v) epi(mbase + v + half * 8, col, c[v]);
}

// ---- weight prep: fp32 -> bf16, transposed to [N][K], xproj N padded to 48 ----
__global__ __launch_bounds__(256) void wprep_kernel(const float* W_in, const float* W_xp,
                                                    const float* W_m1, const float* W_m2,
                                                    __bf16* out) {
  int gid = blockIdx.x * 256 + threadIdx.x;
  int l = gid / WELEMS;
  int r = gid - l * WELEMS;
  float v;
  if (r < 65536) {                       // Wt_in [512][128] <- W_in [l][128][512]
    int n = r >> 7, k = r & 127;
    v = W_in[(size_t)l * 65536 + k * 512 + n];
  } else if (r < 90112) {                // Wt_xp [2][48][256] <- W_xproj [l][2][256][40]
    int r2 = r - 65536;
    int d = r2 / 12288, r3 = r2 - d * 12288;
    int n = r3 >> 8, k = r3 & 255;
    v = (n < 40) ? W_xp[((size_t)(l * 2 + d) * 256 + k) * 40 + n] : 0.0f;
  } else if (r < 352256) {               // Wt_m1 [512][512] <- W_mlp1 [l][512][512]
    int r3 = r - 90112;
    int n = r3 >> 9, k = r3 & 511;
    v = W_m1[(size_t)l * 262144 + k * 512 + n];
  } else {                               // Wt_m2 [128][512] <- W_mlp2 [l][512][128]
    int r4 = r - 352256;
    int n = r4 >> 9, k = r4 & 511;
    v = W_m2[(size_t)l * 65536 + k * 128 + n];
  }
  out[gid] = (__bf16)v;
}

struct LayerArgs {
  const float* x_in;
  float* x_out;
  const float* ln_scale;
  const float* ln_bias;
  const float* b_in;
  const float* conv_w;
  const float* conv_b;
  const float* W_dt;
  const float* b_dt;
  const float* D_skip;
  const float* b_mlp1;
  const float* b_mlp2;
  const __bf16* wts;
  int layer;
};

__global__ __launch_bounds__(256) void layer_kernel(LayerArgs A) {
  extern __shared__ unsigned char smem[];
  const int tid = threadIdx.x;
  const int lane = tid & 31;
  const int wid = tid >> 5;
  const int token0 = blockIdx.x * TM;
  const int l = A.layer;

  __bf16* H    = (__bf16*)(smem + SO_H);
  float*  CW   = (float*)(smem + SO_CW);
  float*  CB   = (float*)(smem + SO_CB);
  __bf16* XA   = (__bf16*)(smem + SO_XA);
  __bf16* GZ   = (__bf16*)(smem + SO_GZ);
  __bf16* CAT  = (__bf16*)(smem + SO_CAT);
  float*  DBC  = (float*)(smem + SO_DBC);
  __bf16* WDT  = (__bf16*)(smem + SO_WDT);
  float*  BDT  = (float*)(smem + SO_BDT);
  float*  DSK  = (float*)(smem + SO_DSK);
  __bf16* M2   = (__bf16*)(smem + SO_M2);
  float*  OUTS = (float*)(smem + SO_OUT);

  // ---- Stage A: LayerNorm -> H (bf16); also stage conv scale/bias for B's epilogue ----
  {
    int r = tid >> 3, j = tid & 7;
    const float* xr = A.x_in + (size_t)(token0 + r) * 128 + j * 16;
    float v[16];
    float s0 = 0.0f, s1 = 0.0f;
#pragma unroll
    for (int t = 0; t < 16; ++t) { v[t] = xr[t]; s0 += v[t]; s1 += v[t] * v[t]; }
    s0 += __shfl_xor(s0, 1); s0 += __shfl_xor(s0, 2); s0 += __shfl_xor(s0, 4);
    s1 += __shfl_xor(s1, 1); s1 += __shfl_xor(s1, 2); s1 += __shfl_xor(s1, 4);
    float mu = s0 * (1.0f / 128.0f);
    float var = s1 * (1.0f / 128.0f) - mu * mu;
    float rstd = __builtin_amdgcn_rsqf(var + 1e-6f);
    const float* lns = A.ln_scale + l * 128;
    const float* lnb = A.ln_bias + l * 128;
#pragma unroll
    for (int t = 0; t < 16; ++t) {
      int c = j * 16 + t;
      H[r * 128 + c] = (__bf16)((v[t] - mu) * rstd * lns[c] + lnb[c]);
    }
    for (int i = tid; i < 512; i += 256) {
      int d = i >> 8, c = i & 255;
      CW[i] = A.conv_w[((size_t)(l * 2 + d) * 256 + c) * 4 + 3];
      CB[i] = A.conv_b[(size_t)(l * 2 + d) * 256 + c];
    }
  }
  __syncthreads();

  // ---- Stage B: xz = H @ W_in + b_in; fused epilogues.
  //      ntiles 0..15  (xin cols)  -> XA[d] = gelu(xin*cw+cb)
  //      ntiles 16..31 (z cols)    -> GZ = gelu(z)
  //      Two loops so the column branch is compile-time (no EXEC juggling). ----
  {
    const __bf16* WIN = A.wts + OFF_WIN;
    const float* bin = A.b_in + l * 512;
    auto epiX = [&](int row, int col, float acc) {
      float val = acc + bin[col];
      XA[row * 256 + col]        = (__bf16)gelu_f(val * CW[col] + CB[col]);
      XA[8192 + row * 256 + col] = (__bf16)gelu_f(val * CW[256 + col] + CB[256 + col]);
    };
    for (int t = wid; t < 32; t += 8) {         // mtile = t&1, ntile = t>>1 in 0..15
      int mtile = t & 1, ntile = t >> 1;
      gemm_tile<4>(H, 128, mtile * 16, WIN, 128, ntile * 16, lane, epiX);
    }
    auto epiZ = [&](int row, int col, float acc) {
      GZ[row * 256 + (col - 256)] = (__bf16)gelu_f(acc + bin[col]);
    };
    for (int t = wid; t < 32; t += 8) {         // ntile = 16 + (t>>1)
      int mtile = t & 1, ntile = 16 + (t >> 1);
      gemm_tile<4>(H, 128, mtile * 16, WIN, 128, ntile * 16, lane, epiZ);
    }
  }
  __syncthreads();

  // ---- Stage D: dbc = xa @ W_xproj (M=32,K=256,N=48 padded) x2; stage W_dt/b_dt/D_skip ----
  for (int i = tid; i < 4096; i += 256) WDT[i] = (__bf16)A.W_dt[(size_t)l * 4096 + i];
  for (int i = tid; i < 512; i += 256) {
    BDT[i] = A.b_dt[(size_t)l * 512 + i];
    DSK[i] = A.D_skip[(size_t)l * 512 + i];
  }
  {
    const __bf16* WXP = A.wts + OFF_WXP;
    for (int t = wid; t < 12; t += 8) {
      int d = t / 6, rm = t % 6, mtile = rm / 3, ntile = rm % 3;
      float* DBCd = DBC + d * 1536;
      auto epi = [&](int row, int col, float acc) { DBCd[row * 48 + col] = acc; };
      gemm_tile<8>(XA + d * 8192, 256, mtile * 16, WXP + d * 12288, 256, ntile * 16, lane, epi);
    }
  }
  __syncthreads();

  // ---- Stage E: dt = softplus(dtr@W_dt + b_dt); BC = Bm.Cm; y = dt*xa*BC + D*xa; CAT = y*gz ----
  {
    int idx = tid >> 2, sub = tid & 3;
    int d = idx >> 5, r = idx & 31;
    const float* dbcrow = DBC + d * 1536 + r * 48;
    float bc = 0.0f;
#pragma unroll
    for (int n = 0; n < 4; ++n) bc += dbcrow[8 + sub * 4 + n] * dbcrow[24 + sub * 4 + n];
    bc += __shfl_xor(bc, 1);
    bc += __shfl_xor(bc, 2);
    float dtr[8];
#pragma unroll
    for (int rr = 0; rr < 8; ++rr) dtr[rr] = dbcrow[rr];
    const __bf16* wdtd = WDT + d * 2048;
    const int e0 = sub * 64;
    float acc[64];
#pragma unroll
    for (int k = 0; k < 64; ++k) acc[k] = BDT[d * 256 + e0 + k];
#pragma unroll
    for (int rr = 0; rr < 8; ++rr) {
      float w = dtr[rr];
#pragma unroll
      for (int k = 0; k < 64; ++k) acc[k] += w * (float)wdtd[rr * 256 + e0 + k];
    }
#pragma unroll
    for (int k = 0; k < 64; ++k) {
      int e = e0 + k;
      float dtv = softplus_f(acc[k]);
      float xav = (float)XA[d * 8192 + r * 256 + e];
      float y = dtv * xav * bc + DSK[d * 256 + e] * xav;
      CAT[r * 512 + d * 256 + e] = (__bf16)(y * (float)GZ[r * 256 + e]);
    }
  }
  __syncthreads();

  // ---- Stage F: mlp1 (M=32,K=512,N=512), fused epilogue -> M2 = bf16(gelu(acc + b_mlp1)) ----
  {
    const __bf16* WM1 = A.wts + OFF_WM1;
    const float* bm1 = A.b_mlp1 + l * 512;
    auto epi = [&](int row, int col, float acc) {
      M2[row * 512 + col] = (__bf16)gelu_f(acc + bm1[col]);
    };
    for (int t = wid; t < 64; t += 8) {
      int mtile = t & 1, ntile = t >> 1;
      gemm_tile<16>(CAT, 512, mtile * 16, WM1, 512, ntile * 16, lane, epi);
    }
  }
  __syncthreads();

  // ---- Stage H: mlp2 (M=32,K=512,N=128) -> OUT = acc + b_mlp2 ----
  {
    const __bf16* WM2 = A.wts + OFF_WM2;
    const float* bm2 = A.b_mlp2 + l * 128;
    auto epi = [&](int row, int col, float acc) { OUTS[row * 128 + col] = acc + bm2[col]; };
    for (int t = wid; t < 16; t += 8) {
      int mtile = t & 1, ntile = t >> 1;
      gemm_tile<16>(M2, 512, mtile * 16, WM2, 512, ntile * 16, lane, epi);
    }
  }
  __syncthreads();

  // ---- Stage I: residual ----
  for (int i = tid; i < TM * 128; i += 256) {
    int r = i >> 7, c = i & 127;
    size_t g = (size_t)(token0 + r) * 128 + c;
    A.x_out[g] = A.x_in[g] + OUTS[i];
  }
}

// ---- final LayerNorm (in-place safe: each element read+written by same thread) ----
__global__ __launch_bounds__(256) void lnf_kernel(const float* xin, float* xout,
                                                  const float* s, const float* b) {
  int tid = threadIdx.x;
  int r = blockIdx.x * TM + (tid >> 3), j = tid & 7;
  const float* xr = xin + (size_t)r * 128 + j * 16;
  float v[16];
  float s0 = 0.0f, s1 = 0.0f;
#pragma unroll
  for (int t = 0; t < 16; ++t) { v[t] = xr[t]; s0 += v[t]; s1 += v[t] * v[t]; }
  s0 += __shfl_xor(s0, 1); s0 += __shfl_xor(s0, 2); s0 += __shfl_xor(s0, 4);
  s1 += __shfl_xor(s1, 1); s1 += __shfl_xor(s1, 2); s1 += __shfl_xor(s1, 4);
  float mu = s0 * (1.0f / 128.0f);
  float var = s1 * (1.0f / 128.0f) - mu * mu;
  float rstd = __builtin_amdgcn_rsqf(var + 1e-6f);
  float* orow = xout + (size_t)r * 128 + j * 16;
#pragma unroll
  for (int t = 0; t < 16; ++t) {
    int c = j * 16 + t;
    orow[t] = (v[t] - mu) * rstd * s[c] + b[c];
  }
}

extern "C" void kernel_launch(void* const* d_in, const int* in_sizes, int n_in,
                              void* d_out, int out_size, void* d_ws, size_t ws_size,
                              hipStream_t stream) {
  (void)in_sizes; (void)n_in; (void)out_size; (void)ws_size;
  const float* x      = (const float*)d_in[0];
  const float* ln_s   = (const float*)d_in[1];
  const float* ln_b   = (const float*)d_in[2];
  const float* W_in   = (const float*)d_in[3];
  const float* b_in   = (const float*)d_in[4];
  const float* conv_w = (const float*)d_in[5];
  const float* conv_b = (const float*)d_in[6];
  const float* W_xp   = (const float*)d_in[7];
  const float* W_dt   = (const float*)d_in[8];
  const float* b_dt   = (const float*)d_in[9];
  const float* D_sk   = (const float*)d_in[10];
  const float* W_m1   = (const float*)d_in[11];
  const float* b_m1   = (const float*)d_in[12];
  const float* W_m2   = (const float*)d_in[13];
  const float* b_m2   = (const float*)d_in[14];
  const float* lnf_s  = (const float*)d_in[15];
  const float* lnf_b  = (const float*)d_in[16];

  float* xping = (float*)d_ws;                                   // 32 MB ping buffer
  __bf16* warena = (__bf16*)((char*)d_ws + 33554432);            // bf16 weight arena
  float* outf = (float*)d_out;

  wprep_kernel<<<(NLAY * WELEMS) / 256, 256, 0, stream>>>(W_in, W_xp, W_m1, W_m2, warena);

  const float* cur = x;
  for (int l = 0; l < NLAY; ++l) {
    float* dst = (l & 1) ? outf : xping;
    LayerArgs a;
    a.x_in = cur; a.x_out = dst;
    a.ln_scale = ln_s; a.ln_bias = ln_b;
    a.b_in = b_in; a.conv_w = conv_w; a.conv_b = conv_b;
    a.W_dt = W_dt; a.b_dt = b_dt; a.D_skip = D_sk;
    a.b_mlp1 = b_m1; a.b_mlp2 = b_m2;
    a.wts = warena + (size_t)l * WELEMS;
    a.layer = l;
    layer_kernel<<<NBLOCKS, 256, SMEM_BYTES, stream>>>(a);
    cur = dst;
  }
  lnf_kernel<<<NBLOCKS, 256, 0, stream>>>(cur, outf, lnf_s, lnf_b);
}
